// SGRUCell_89696097009938
// MI455X (gfx1250) — matrix operand order repaired and checked by reading, not compiled, not run on hardware
//
#include <hip/hip_runtime.h>
#include <hip/hip_bf16.h>

// Problem constants (match reference).
#define T_STEPS 32
#define BATCH   8
#define IDIM    512
#define HDIM    512
#define H3      1536
#define CLIPV   1.0f

typedef __attribute__((ext_vector_type(16))) _Float16 v16h;
typedef __attribute__((ext_vector_type(8)))  _Float16 v8h;
typedef __attribute__((ext_vector_type(8)))  float    v8f;

static __device__ __forceinline__ float sigmoidf_(float x) {
    return 1.0f / (1.0f + __expf(-x));
}
static __device__ __forceinline__ float softshrinkf_(float x) {
    return (x > 0.5f) ? (x - 0.5f) : ((x < -0.5f) ? (x + 0.5f) : 0.0f);
}

// ---------------------------------------------------------------------------
// Wave-level 16x16 WMMA tile GEMM: D = A(16xK, row-major f16) * B(Kx16).
// B is stored K-major with N contiguous (ldb = full N width), so each lane's
// fragment is 16 scalar loads strided by ldb.
// A fragment layout (ISA 7.12.2, 16-bit A 16x32): lane m (0-15) / m+16 holds
// row m; halves 0..7 = K {0..7 | 8..15}, halves 8..15 = K {16..23 | 24..31}.
// B fragment: lane n (0-15) / n+16 holds column n; halves same K order as A.
// ---------------------------------------------------------------------------
__device__ __forceinline__ v8f wmma_rowA_kmajB(const _Float16* __restrict__ A,
                                               int lda, int arows,
                                               const _Float16* __restrict__ Bt,
                                               int ldb, int K) {
    const int lane = threadIdx.x & 31;
    const int m    = lane & 15;
    const int hi   = lane >> 4;
    v8f acc = {};
    for (int k0 = 0; k0 < K; k0 += 32) {
        v8h a0 = {}, a1 = {};
        if (m < arows) {
            const _Float16* ap = A + (size_t)m * lda + k0 + hi * 8;
            a0 = *(const v8h*)(ap);
            a1 = *(const v8h*)(ap + 16);
        }
        v16h a = __builtin_shufflevector(a0, a1, 0, 1, 2, 3, 4, 5, 6, 7,
                                         8, 9, 10, 11, 12, 13, 14, 15);
        v16h b;
#pragma unroll
        for (int s = 0; s < 16; ++s)
            b[s] = Bt[(size_t)(k0 + (s < 8 ? hi * 8 + s : 16 + hi * 8 + s - 8)) * ldb + m];
        acc = __builtin_amdgcn_wmma_f32_16x16x32_f16(
            /*neg_a=*/false, a, /*neg_b=*/false, b,
            /*c_mod=*/(short)0, acc, /*reuse_a=*/false, /*reuse_b=*/false);
    }
    return acc;
}

// ---------------------------------------------------------------------------
// Prep: weight-norm both matrices; write f16 transposed (K-major) packs for
// WMMA B operands; compute dU clip bounds up/lo from normalized Wr rows.
// One wave per output row (3072 rows total).
// ---------------------------------------------------------------------------
__global__ void wn_prep_kernel(const float* __restrict__ x2h_v,
                               const float* __restrict__ x2h_g,
                               const float* __restrict__ h2h_v,
                               const float* __restrict__ h2h_g,
                               const float* __restrict__ alpha,
                               _Float16* __restrict__ WxT,
                               _Float16* __restrict__ WhT,
                               float* __restrict__ up, float* __restrict__ lo) {
    int wave = (int)((blockIdx.x * blockDim.x + threadIdx.x) >> 5);
    int lane = threadIdx.x & 31;
    if (wave >= 2 * H3) return;
    const bool isX = wave < H3;
    const int  row = isX ? wave : wave - H3;
    const float* v = (isX ? x2h_v : h2h_v) + (size_t)row * HDIM;
    const float  g = (isX ? x2h_g : h2h_g)[row];

    float s = 0.0f;
    for (int c = lane; c < HDIM; c += 32) { float t = v[c]; s += t * t; }
    for (int off = 16; off > 0; off >>= 1) s += __shfl_xor(s, off, 32);
    const float scale = g / sqrtf(s);

    _Float16* Wt = isX ? WxT : WhT;
    const bool isR = (!isX) && (row >= HDIM) && (row < 2 * HDIM);
    const int  i   = row - HDIM;
    for (int c = lane; c < HDIM; c += 32) {
        float w = v[c] * scale;
        Wt[(size_t)c * H3 + row] = (_Float16)w;
        if (isR) {
            float a = alpha[(size_t)i * HDIM + c] + 1e-5f;
            up[(size_t)i * HDIM + c] = fmaxf(CLIPV - w, 0.0f) / a;
            lo[(size_t)i * HDIM + c] = -fmaxf(CLIPV + w, 0.0f) / a;
        }
    }
}

__global__ void f32_to_f16_kernel(const float* __restrict__ src,
                                  _Float16* __restrict__ dst, int n) {
    int idx = blockIdx.x * blockDim.x + threadIdx.x;
    if (idx < n) dst[idx] = (_Float16)src[idx];
}

__global__ void init_state_kernel(const float* __restrict__ h0,
                                  const float* __restrict__ v0,
                                  const float* __restrict__ te0,
                                  float* __restrict__ h, float* __restrict__ v,
                                  float* __restrict__ te,
                                  _Float16* __restrict__ hf16) {
    int e = blockIdx.x * blockDim.x + threadIdx.x;
    if (e < BATCH * HDIM) {
        float hh = h0[e];
        h[e] = hh;
        hf16[e] = (_Float16)hh;
        v[e] = v0[e];
        te[e] = te0[e];
    }
}

// xproj = x @ Wx^T + x2h_b ; M = T*B = 256, K = 512, N = 1536.
__global__ void xproj_kernel(const _Float16* __restrict__ xf16,
                             const _Float16* __restrict__ WxT,
                             const float* __restrict__ x2h_b,
                             float* __restrict__ xproj) {
    const int NT = H3 / 16;  // 96 N tiles
    int wave = (int)((blockIdx.x * blockDim.x + threadIdx.x) >> 5);
    const int lane = threadIdx.x & 31;
    if (wave >= (T_STEPS * BATCH / 16) * NT) return;
    const int mt = wave / NT, nt = wave % NT;
    v8f acc = wmma_rowA_kmajB(xf16 + (size_t)mt * 16 * IDIM, IDIM, 16,
                              WxT + nt * 16, H3, IDIM);
    const int n = lane & 15, hi = lane >> 4;
    const int o = nt * 16 + n;
    const float bias = x2h_b[o];
#pragma unroll
    for (int r = 0; r < 8; ++r) {
        int mrow = mt * 16 + hi * 8 + r;
        xproj[(size_t)mrow * H3 + o] = acc[r] + bias;
    }
}

// hp = h @ Wh^T + h2h_b ; M = 8 (padded to 16), K = 512, N = 1536.
__global__ void hproj_kernel(const _Float16* __restrict__ hf16,
                             const _Float16* __restrict__ WhT,
                             const float* __restrict__ h2h_b,
                             float* __restrict__ hp) {
    int wave = (int)((blockIdx.x * blockDim.x + threadIdx.x) >> 5);
    const int lane = threadIdx.x & 31;
    if (wave >= H3 / 16) return;
    v8f acc = wmma_rowA_kmajB(hf16, HDIM, BATCH, WhT + wave * 16, H3, HDIM);
    const int n = lane & 15, hi = lane >> 4;
    if (hi == 0) {  // only rows 0..7 are real batch rows
        const int o = wave * 16 + n;
        const float bias = h2h_b[o];
#pragma unroll
        for (int r = 0; r < 8; ++r)
            hp[(size_t)r * H3 + o] = acc[r] + bias;
    }
}

// Standalone matvec (only used for t = 0, against dU0):
// m[b,i] = sum_j alpha[i,j] * dU[b,i,j] * h[b,j] ; one wave per (b,i) row.
__global__ void matvec_kernel(const float* __restrict__ dU,
                              const float* __restrict__ alpha,
                              const float* __restrict__ h,
                              float* __restrict__ mvec) {
    int wave = (int)((blockIdx.x * blockDim.x + threadIdx.x) >> 5);
    int lane = threadIdx.x & 31;
    if (wave >= BATCH * HDIM) return;
    const int b = wave >> 9, i = wave & (HDIM - 1);
    const float* du = dU + ((size_t)b * HDIM + i) * HDIM;
    const float* al = alpha + (size_t)i * HDIM;
    const float* hb = h + b * HDIM;
    float s = 0.0f;
    for (int j = lane; j < HDIM; j += 32) s += al[j] * du[j] * hb[j];
    for (int off = 16; off > 0; off >>= 1) s += __shfl_xor(s, off, 32);
    if (lane == 0) mvec[wave] = s;
}

// Small per-step pointwise update: gates, v/h update, mod GEMV, te update.
// Single 512-thread block (B*H = 4096 elements, 8 per thread).
__global__ void __launch_bounds__(512)
step_small_kernel(const float* __restrict__ xproj_t, const float* __restrict__ hp,
                  const float* __restrict__ mvec,
                  const float* __restrict__ h2mod_w,
                  const float* __restrict__ h2mod_b,
                  float* __restrict__ h, float* __restrict__ v,
                  float* __restrict__ te, _Float16* __restrict__ hf16,
                  float* __restrict__ taus, float* __restrict__ outs_t) {
    __shared__ float s_hn[BATCH * HDIM];
    __shared__ float s_mod[32];
    __shared__ float s_taue[BATCH];
    const int tid = threadIdx.x;
    float hold[8];
#pragma unroll
    for (int kk = 0; kk < 8; ++kk) {
        const int e = tid + kk * 512;
        const int b = e >> 9, idx = e & (HDIM - 1);
        const float zx  = xproj_t[(size_t)b * H3 + idx];
        const float zh  = hp[(size_t)b * H3 + idx];
        const float dvx = xproj_t[(size_t)b * H3 + 2 * HDIM + idx];
        const float dvh = hp[(size_t)b * H3 + 2 * HDIM + idx];
        const float z  = sigmoidf_(zx + zh);
        const float dv = dvx + dvh + mvec[e];
        const float vn = (1.0f - z) * v[e] + z * dv;
        const float hn = fmaxf(vn, 0.0f);
        v[e]     = vn;
        hold[kk] = h[e];      // stash OLD h (te update uses it)
        h[e]     = hn;
        hf16[e]  = (_Float16)hn;
        s_hn[e]  = hn;
        outs_t[e] = hn;
    }
    __syncthreads();
    // mod = hn @ h2mod_w^T + h2mod_b  -> (B,4); 32 wave reductions.
    const int wid = tid >> 5, lane = tid & 31;
#pragma unroll
    for (int pass = 0; pass < 2; ++pass) {
        const int pp = wid + pass * 16;
        const int b = pp >> 2, kk = pp & 3;
        float s = 0.0f;
        for (int i2 = lane; i2 < HDIM; i2 += 32)
            s += s_hn[b * HDIM + i2] * h2mod_w[kk * HDIM + i2];
        for (int off = 16; off > 0; off >>= 1) s += __shfl_xor(s, off, 32);
        if (lane == 0) s_mod[pp] = s + h2mod_b[kk];
    }
    __syncthreads();
    if (tid < 32) {
        const int b = tid >> 2, kk = tid & 3;
        const float mval = s_mod[tid];
        const float tv = (kk == 3) ? fmaxf(mval, 0.0f) : sigmoidf_(mval);
        if (kk == 0) s_taue[b] = tv;
        taus[tid] = tv;  // [b*4 + {tau_e,tau_E,tau_U,mU}]
    }
    __syncthreads();
#pragma unroll
    for (int kk = 0; kk < 8; ++kk) {
        const int e = tid + kk * 512;
        const int b = e >> 9;
        const float tau = s_taue[b];
        te[e] = (1.0f - tau) * te[e] + tau * hold[kk];  // uses OLD h
    }
}

// Big per-step elementwise over B*H*H fused with next step's matvec:
// each 256-thread block owns TWO complete (b,i) rows (128 threads x float4
// = 512 j's per row), updates tE/dU in place, and reduces
// m[b,i] = sum_j alpha[i,j] * dU_new[b,i,j] * hn[b,j] for the next step
// (saves re-reading the 8 MB dU tensor).
__global__ void __launch_bounds__(256)
step_big_kernel(const float* __restrict__ hn, const float* __restrict__ te,
                const float* __restrict__ taus,
                const float* __restrict__ modU_w, const float* __restrict__ modU_b,
                const float* __restrict__ lo, const float* __restrict__ up,
                const float* __restrict__ alpha,
                float* __restrict__ dU, float* __restrict__ tE,
                float* __restrict__ mvec) {
    __shared__ float s_part[8];  // one partial per wave (4 waves per row)
    const int tid = threadIdx.x;
    const int r   = tid >> 7;                    // local row 0/1
    const int jj  = (tid & 127) * 4;             // j offset within row
    const int row = blockIdx.x * 2 + r;          // global (b,i) row, < 4096
    const int b = row >> 9;
    const int i = row & (HDIM - 1);
    const size_t base = (size_t)row * HDIM + jj;

    const float tauE = taus[b * 4 + 1];
    const float tauU = taus[b * 4 + 2];
    const float mU   = taus[b * 4 + 3];
    const float hni = hn[b * HDIM + i];
    const float tei = te[b * HDIM + i];
    const float4 hnj = *(const float4*)(hn + b * HDIM + jj);
    const float4 tej = *(const float4*)(te + b * HDIM + jj);
    const float4 tE4 = *(const float4*)(tE + base);
    const float4 dU4 = *(const float4*)(dU + base);
    const size_t ij = (size_t)i * HDIM + jj;
    const float4 w4  = *(const float4*)(modU_w + ij);
    const float4 mb4 = *(const float4*)(modU_b + ij);
    const float4 lo4 = *(const float4*)(lo + ij);
    const float4 up4 = *(const float4*)(up + ij);
    const float4 al4 = *(const float4*)(alpha + ij);

    float r_tE[4], r_dU[4];
    const float hnjv[4] = {hnj.x, hnj.y, hnj.z, hnj.w};
    const float tejv[4] = {tej.x, tej.y, tej.z, tej.w};
    const float tEv[4]  = {tE4.x, tE4.y, tE4.z, tE4.w};
    const float dUv[4]  = {dU4.x, dU4.y, dU4.z, dU4.w};
    const float wv[4]   = {w4.x, w4.y, w4.z, w4.w};
    const float mbv[4]  = {mb4.x, mb4.y, mb4.z, mb4.w};
    const float lov[4]  = {lo4.x, lo4.y, lo4.z, lo4.w};
    const float upv[4]  = {up4.x, up4.y, up4.z, up4.w};
    const float alv[4]  = {al4.x, al4.y, al4.z, al4.w};
    float part = 0.0f;
#pragma unroll
    for (int c = 0; c < 4; ++c) {
        const float outer = hni * tejv[c] - tei * hnjv[c];
        const float tEn = (1.0f - tauE) * tEv[c] + tauE * outer;
        const float sh  = softshrinkf_(mU * wv[c] + mbv[c]);
        float dUn = (1.0f - tauU) * dUv[c] + tauU * sh * tEn;
        dUn = fminf(fmaxf(dUn, lov[c]), upv[c]);
        r_tE[c] = tEn;
        r_dU[c] = dUn;
        part += alv[c] * dUn * hnjv[c];  // next step's (alpha*dU)@h
    }
    *(float4*)(tE + base) = make_float4(r_tE[0], r_tE[1], r_tE[2], r_tE[3]);
    *(float4*)(dU + base) = make_float4(r_dU[0], r_dU[1], r_dU[2], r_dU[3]);

    // Reduce `part` over the 128 threads (4 waves) covering this row.
    const int lane = tid & 31;
    for (int off = 16; off > 0; off >>= 1) part += __shfl_xor(part, off, 32);
    if (lane == 0) s_part[tid >> 5] = part;
    __syncthreads();
    if (tid < 2) {
        const int w0 = tid * 4;
        mvec[blockIdx.x * 2 + tid] =
            s_part[w0] + s_part[w0 + 1] + s_part[w0 + 2] + s_part[w0 + 3];
    }
}

__global__ void finalize_kernel(const float* __restrict__ h,
                                const float* __restrict__ v,
                                const float* __restrict__ te,
                                float* __restrict__ out) {
    int e = blockIdx.x * blockDim.x + threadIdx.x;
    if (e < BATCH * HDIM) {
        out[e] = v[e];                    // v
        out[4096 + e] = h[e];             // h
        out[2105344 + e] = te[e];         // te
    }
}

extern "C" void kernel_launch(void* const* d_in, const int* in_sizes, int n_in,
                              void* d_out, int out_size, void* d_ws, size_t ws_size,
                              hipStream_t stream) {
    const float* x       = (const float*)d_in[0];
    const float* h0      = (const float*)d_in[1];
    const float* v0      = (const float*)d_in[2];
    const float* dU0     = (const float*)d_in[3];
    const float* te0     = (const float*)d_in[4];
    const float* tE0     = (const float*)d_in[5];
    const float* x2h_v   = (const float*)d_in[6];
    const float* x2h_g   = (const float*)d_in[7];
    const float* x2h_b   = (const float*)d_in[8];
    const float* h2h_v   = (const float*)d_in[9];
    const float* h2h_g   = (const float*)d_in[10];
    const float* h2h_b   = (const float*)d_in[11];
    const float* alpha   = (const float*)d_in[12];
    const float* h2mod_w = (const float*)d_in[13];
    const float* h2mod_b = (const float*)d_in[14];
    const float* modU_w  = (const float*)d_in[15];
    const float* modU_b  = (const float*)d_in[16];

    // d_out layout: v | h | dU | te | tE | outs  (floats)
    float* out      = (float*)d_out;
    float* out_dU   = out + 8192;
    float* out_tE   = out + 2109440;
    float* out_outs = out + 4206592;

    // Workspace carve (256B aligned chunks).
    char* ws = (char*)d_ws;
    size_t off = 0;
    auto carve = [&](size_t bytes) -> char* {
        char* p = ws + off;
        off += (bytes + 255) & ~(size_t)255;
        return p;
    };
    _Float16* WxT  = (_Float16*)carve((size_t)IDIM * H3 * 2);
    _Float16* WhT  = (_Float16*)carve((size_t)HDIM * H3 * 2);
    _Float16* xf16 = (_Float16*)carve((size_t)T_STEPS * BATCH * IDIM * 2);
    float* xproj   = (float*)carve((size_t)T_STEPS * BATCH * H3 * 4);
    float* up      = (float*)carve((size_t)HDIM * HDIM * 4);
    float* lo      = (float*)carve((size_t)HDIM * HDIM * 4);
    float* hp      = (float*)carve((size_t)BATCH * H3 * 4);
    float* mvec    = (float*)carve((size_t)BATCH * HDIM * 4);
    float* h_st    = (float*)carve((size_t)BATCH * HDIM * 4);
    float* v_st    = (float*)carve((size_t)BATCH * HDIM * 4);
    float* te_st   = (float*)carve((size_t)BATCH * HDIM * 4);
    _Float16* hf16 = (_Float16*)carve((size_t)BATCH * HDIM * 2);
    float* taus    = (float*)carve(32 * 4);
    (void)in_sizes; (void)n_in; (void)out_size; (void)ws_size;

    // ---- prep ----
    wn_prep_kernel<<<384, 256, 0, stream>>>(x2h_v, x2h_g, h2h_v, h2h_g, alpha,
                                            WxT, WhT, up, lo);
    f32_to_f16_kernel<<<512, 256, 0, stream>>>(x, xf16, T_STEPS * BATCH * IDIM);
    init_state_kernel<<<16, 256, 0, stream>>>(h0, v0, te0, h_st, v_st, te_st, hf16);
    hipMemcpyAsync(out_dU, dU0, (size_t)BATCH * HDIM * HDIM * 4,
                   hipMemcpyDeviceToDevice, stream);
    hipMemcpyAsync(out_tE, tE0, (size_t)BATCH * HDIM * HDIM * 4,
                   hipMemcpyDeviceToDevice, stream);
    xproj_kernel<<<192, 256, 0, stream>>>(xf16, WxT, x2h_b, xproj);

    // mvec for t=0 from (dU0, h0); subsequent steps get it fused from
    // step_big (which already holds the fresh dU in registers).
    matvec_kernel<<<512, 256, 0, stream>>>(out_dU, alpha, h_st, mvec);

    // ---- sequential scan (dU/tE state lives in d_out) ----
    for (int t = 0; t < T_STEPS; ++t) {
        hproj_kernel<<<12, 256, 0, stream>>>(hf16, WhT, h2h_b, hp);
        step_small_kernel<<<1, 512, 0, stream>>>(
            xproj + (size_t)t * BATCH * H3, hp, mvec, h2mod_w, h2mod_b,
            h_st, v_st, te_st, hf16, taus, out_outs + (size_t)t * BATCH * HDIM);
        step_big_kernel<<<2048, 256, 0, stream>>>(h_st, te_st, taus, modU_w,
                                                  modU_b, lo, up, alpha,
                                                  out_dU, out_tE, mvec);
    }
    finalize_kernel<<<16, 256, 0, stream>>>(h_st, v_st, te_st, out);
}
